// unitedCLLoss_3728031613431
// MI455X (gfx1250) — matrix operand order repaired and checked by reading, not compiled, not run on hardware
//
#include <hip/hip_runtime.h>
#include <hip/hip_bf16.h>

typedef __attribute__((ext_vector_type(16))) _Float16 v16h;
typedef __attribute__((ext_vector_type(8)))  _Float16 v8h;
typedef __attribute__((ext_vector_type(8)))  float    v8f;

#define N_TOT   8192
#define BSZ     4096
#define D       256
// log2(e)/0.07
#define SCALE_L2E 20.60992915555662f
#define LN2       0.6931471805599453f

// ---------------------------------------------------------------------------
// Kernel 1: convert features (f32, [bsz, views, d]) -> contrast-ordered f16
// panel Ah[8192][256]; also expand labels and zero the S/P accumulators.
// ---------------------------------------------------------------------------
__global__ void supcon_prep(const float* __restrict__ feats,
                            const int*   __restrict__ labels,
                            _Float16* __restrict__ Ah,
                            int*      __restrict__ labc,
                            float*    __restrict__ S,
                            float*    __restrict__ P) {
  int gid = blockIdx.x * blockDim.x + threadIdx.x;   // 0 .. 8192*256-1
  int i = gid >> 8;          // contrast row 0..8191
  int k = gid & 255;
  int b = i & (BSZ - 1);     // batch
  int v = i >> 12;           // view
  Ah[gid] = (_Float16)feats[((size_t)(b * 2 + v)) * D + k];
  if (gid < N_TOT) {
    labc[gid] = labels[gid & (BSZ - 1)];
    S[gid] = 0.0f;
    P[gid] = 0.0f;
  }
}

// ---------------------------------------------------------------------------
// Async stage: copy one 64-col x 256-half B tile (32 KB) global -> LDS using
// gfx1250 GLOBAL_LOAD_ASYNC_TO_LDS_B128 (tracked by ASYNCcnt, overlaps WMMA).
// Tile layout in LDS: [col_local][k] row-major, 512 B per column row.
// ---------------------------------------------------------------------------
__device__ __forceinline__ void stage_tile(const _Float16* __restrict__ Ah,
                                           int colTile,
                                           _Float16* bufPtr) {
  const char* gbase = (const char*)Ah;
  uint32_t lds0 = (uint32_t)(size_t)bufPtr;   // addr[31:0] == LDS byte offset
  int t = threadIdx.x;
#pragma unroll
  for (int i = 0; i < 8; ++i) {
    int c = t + i * 256;                 // chunk id, 2048 x 16B chunks
    int col_local = c >> 5;              // 32 chunks (512 B) per column
    int kb = (c & 31) * 16;              // byte offset within the column row
    const char* g = gbase + ((size_t)(colTile + col_local) << 9) + kb;
    uint32_t l = lds0 + (uint32_t)(c * 16);
    asm volatile("global_load_async_to_lds_b128 %0, %1, off"
                 :: "v"(l), "v"(g) : "memory");
  }
}

// B fragment (32x16 f16) from LDS: lanes 0-15 hold K 0..15 consecutively,
// lanes 16-31 hold K 16..31 (contiguous k order, same as A). bcol already includes
// the per-lane column and lane-half offsets.
__device__ __forceinline__ v16h ldb(const _Float16* bcol, int kc) {
  const _Float16* p = bcol + kc * 32;
  v8h lo = *(const v8h*)(p);
  v8h hi = *(const v8h*)(p + 8);
  return __builtin_shufflevector(lo, hi, 0, 1, 2, 3, 4, 5, 6, 7,
                                         8, 9, 10, 11, 12, 13, 14, 15);
}

// ---------------------------------------------------------------------------
// Kernel 2: fused Gram-tile WMMA + exp + masked row sums.
// Block: 256 threads = 8 waves. Block tile: 128 rows x 1024 cols.
// B panel double-buffered in LDS via async-to-LDS DMA; B fragments register
// double-buffered; two concurrent accumulator chains per K step.
// ---------------------------------------------------------------------------
__global__ __launch_bounds__(256)
void supcon_gemm(const _Float16* __restrict__ Ah,
                 const int*      __restrict__ labc,
                 float* __restrict__ S,
                 float* __restrict__ P) {
  __shared__ __align__(16) _Float16 Bs[2][64 * 256];   // 2 x 32 KB

  const int lane = threadIdx.x & 31;
  const int wave = threadIdx.x >> 5;
  const int rowBase = blockIdx.x * 128 + wave * 16;   // 16 rows per wave
  const int panel   = blockIdx.y * 1024;              // 1024-col panel
  const int lh = (lane >> 4) & 1;                     // lane half
  const int nl = lane & 15;

  // --- A fragments (16x32 f16): lanes 0-15 K=0..15,
  //     lanes 16-31 K=16..31 (offset +16); cached for all 8 k-chunks.
  v16h a[8];
  {
    const _Float16* arow = Ah + (size_t)(rowBase + nl) * D + lh * 16;
#pragma unroll
    for (int kc = 0; kc < 8; ++kc) {
      const _Float16* p = arow + kc * 32;
      v8h lo = *(const v8h*)(p);
      v8h hi = *(const v8h*)(p + 8);
      a[kc] = __builtin_shufflevector(lo, hi, 0, 1, 2, 3, 4, 5, 6, 7,
                                              8, 9, 10, 11, 12, 13, 14, 15);
    }
  }

  // Row labels for this wave's 16 rows (C layout: lanes 0-15 -> M=r,
  // lanes 16-31 -> M=r+8).
  int lr[8];
#pragma unroll
  for (int r = 0; r < 8; ++r) lr[r] = labc[rowBase + r + lh * 8];

  float Sr[8] = {0.f, 0.f, 0.f, 0.f, 0.f, 0.f, 0.f, 0.f};
  float Pr[8] = {0.f, 0.f, 0.f, 0.f, 0.f, 0.f, 0.f, 0.f};

  // Prologue: stage tile 0 into buffer 0.
  stage_tile(Ah, panel, Bs[0]);
  asm volatile("s_wait_asynccnt 0x0" ::: "memory");
  __syncthreads();

  for (int ct = 0; ct < 16; ++ct) {          // 16 col-tiles of 64
    const _Float16* cur = Bs[ct & 1];
    // Kick off the DMA for the next tile while we compute on this one.
    if (ct + 1 < 16) stage_tile(Ah, panel + (ct + 1) * 64, Bs[(ct + 1) & 1]);

#pragma unroll
    for (int jp = 0; jp < 2; ++jp) {         // 2 pairs of 16-col sub-tiles
      const int colBase = panel + ct * 64 + jp * 32;
      const _Float16* bcol0 = cur + (size_t)(jp * 32 + nl) * D + lh * 16;
      const _Float16* bcol1 = bcol0 + 16 * D;

      v8f acc0 = {};
      v8f acc1 = {};
      // Register double-buffer: fragments for kc are loaded before the
      // WMMAs of kc-1 retire, so DScnt waits never fully drain.
      v16h b0 = ldb(bcol0, 0);
      v16h b1 = ldb(bcol1, 0);
#pragma unroll
      for (int kc = 0; kc < 8; ++kc) {
        v16h n0, n1;
        if (kc < 7) {
          n0 = ldb(bcol0, kc + 1);
          n1 = ldb(bcol1, kc + 1);
        }
        acc0 = __builtin_amdgcn_wmma_f32_16x16x32_f16(
            false, a[kc], false, b0, (short)0, acc0, false, false);
        acc1 = __builtin_amdgcn_wmma_f32_16x16x32_f16(
            false, a[kc], false, b1, (short)0, acc1, false, false);
        if (kc < 7) { b0 = n0; b1 = n1; }
      }

      // --- exp + masks + row-sum accumulation for both sub-tiles
#pragma unroll
      for (int half = 0; half < 2; ++half) {
        const v8f& acc = half ? acc1 : acc0;
        const int jcol = colBase + half * 16 + nl;
        const int lj = labc[jcol];
#pragma unroll
        for (int r = 0; r < 8; ++r) {
          const int irow = rowBase + r + lh * 8;
          float sim = __builtin_amdgcn_exp2f(acc[r] * SCALE_L2E);
          float s = (irow != jcol) ? sim : 0.0f;   // drop self-contrast
          Sr[r] += s;
          Pr[r] += (lr[r] == lj) ? s : 0.0f;
        }
      }
    }

    // Next buffer fully landed + everyone done reading current buffer.
    asm volatile("s_wait_asynccnt 0x0" ::: "memory");
    __syncthreads();
  }

  // --- reduce across the 16 lanes of each half (xor 1,2,4,8), then atomics.
#pragma unroll
  for (int r = 0; r < 8; ++r) {
    float s = Sr[r];
    float p = Pr[r];
#pragma unroll
    for (int m = 8; m >= 1; m >>= 1) {
      s += __shfl_xor(s, m, 32);
      p += __shfl_xor(p, m, 32);
    }
    if (nl == 0) {
      const int irow = rowBase + r + lh * 8;
      atomicAdd(&S[irow], s);
      atomicAdd(&P[irow], p);
    }
  }
}

// ---------------------------------------------------------------------------
// Kernel 3: loss = -(1/n) * sum_i log((P_i + eps*S_i) / S_i)
// ---------------------------------------------------------------------------
__global__ void supcon_finalize(const float* __restrict__ S,
                                const float* __restrict__ P,
                                float* __restrict__ out) {
  float acc = 0.0f;
  for (int i = threadIdx.x; i < N_TOT; i += 256) {
    float s = S[i];
    float p = P[i];
    float ratio = (p + 1e-7f * s) / s;
    acc += LN2 * __builtin_amdgcn_logf(ratio);   // v_log_f32 is log2
  }
#pragma unroll
  for (int m = 16; m >= 1; m >>= 1) acc += __shfl_xor(acc, m, 32);
  __shared__ float wsum[8];
  if ((threadIdx.x & 31) == 0) wsum[threadIdx.x >> 5] = acc;
  __syncthreads();
  if (threadIdx.x == 0) {
    float t = 0.0f;
#pragma unroll
    for (int w = 0; w < 8; ++w) t += wsum[w];
    out[0] = -t / (float)N_TOT;
  }
}

// ---------------------------------------------------------------------------
extern "C" void kernel_launch(void* const* d_in, const int* in_sizes, int n_in,
                              void* d_out, int out_size, void* d_ws, size_t ws_size,
                              hipStream_t stream) {
  const float* feats  = (const float*)d_in[0];   // [4096, 2, 256] f32
  const int*   labels = (const int*)d_in[1];     // [4096] i32
  float* out = (float*)d_out;

  char* ws = (char*)d_ws;
  _Float16* Ah   = (_Float16*)ws;                                  // 4 MiB
  int*      labc = (int*)(ws + (size_t)N_TOT * D * sizeof(_Float16));
  float*    S    = (float*)((char*)labc + (size_t)N_TOT * sizeof(int));
  float*    P    = S + N_TOT;

  // 1) convert + init (8192*256 elements / 256 threads)
  supcon_prep<<<(N_TOT * D) / 256, 256, 0, stream>>>(feats, labels, Ah, labc, S, P);

  // 2) fused WMMA gram + exp + masked row sums (async-LDS double buffered)
  dim3 grid(N_TOT / 128, N_TOT / 1024);   // (64, 8)
  supcon_gemm<<<grid, 256, 0, stream>>>(Ah, labc, S, P);

  // 3) scalar loss
  supcon_finalize<<<1, 256, 0, stream>>>(S, P, out);
}
